// SimpleSNN_11416023073066
// MI455X (gfx1250) — matrix operand order repaired and checked by reading, not compiled, not run on hardware
//
#include <hip/hip_runtime.h>

// ---------------------------------------------------------------------------
// SNN forward: emb-gather -> fc1 (WMMA) -> LIF scan -> fc2 (WMMA, +bias)
// Shapes: B=8 T=256 D=512 H=1024 V=50257, M = B*T = 2048
// A tiles move via the Tensor Data Mover (TDM) global->LDS; B tiles are
// converted f32->f16 in-flight through VGPRs (TDM cannot convert).
// ---------------------------------------------------------------------------

typedef __attribute__((ext_vector_type(16))) _Float16 v16h;
typedef __attribute__((ext_vector_type(8)))  float    v8f;
typedef __attribute__((ext_vector_type(4)))  unsigned u32x4;
typedef __attribute__((ext_vector_type(8)))  int      i32x8;
typedef __attribute__((ext_vector_type(4)))  int      i32x4;

#define BM 128
#define BN 128
#define BK 32
#define SAK 40   // padded LDS row stride (f16): 64 B row + 16 B pad (TDM pad)
#define SBK 40   // padded LDS row stride (f16) for sBt

union F16x16 { v16h v; float4 f4[2]; };
union F16x8  { float4 f4; _Float16 h[8]; };

// ---------------------------------------------------------------------------
// TDM descriptor per CDNA5 ISA:
//  group0: [1:0]=count=1, [63:32]=lds_addr, [120:64]=global_addr, [127:126]=2
//  group1: data_size=1(2B), pad_enable, pad_interval=3 (16 DW = 64 B row),
//          pad_amount=3 (4 DW = 16 B)  -> LDS row stride 80 B == SAK
//          tensor_dim0=K, tensor_dim1=M, tile_dim0=BK(32), tile_dim1=BM(128),
//          tensor_dim0_stride=K
// 2D tile -> groups 2/3 zero.
// ---------------------------------------------------------------------------
template <int K>
__device__ __forceinline__ void tdm_load_a_tile(const _Float16* gsrc,
                                                const _Float16* ldsDst, int M) {
    const unsigned lds = (unsigned)(uintptr_t)ldsDst;  // addr[31:0] = LDS offset
    const unsigned long long ga = (unsigned long long)(uintptr_t)gsrc;

    u32x4 g0;
    g0[0] = 1u;                                   // count=1 (valid user D#)
    g0[1] = lds;                                  // lds_addr
    g0[2] = (unsigned)(ga & 0xFFFFFFFFull);       // global_addr[31:0]
    g0[3] = (unsigned)((ga >> 32) & 0x01FFFFFFull)// global_addr[56:32]
          | (2u << 30);                           // type = 2 ("image")

    i32x8 g1;
    g1[0] = (1 << 16)        // data_size = 1 -> 2 bytes
          | (1 << 20)        // pad_enable
          | (3 << 22)        // pad_interval = 3 -> 16 DWORDs (one 64 B row)
          | (3 << 25);       // pad_amount   = 3 -> 4 DWORDs (16 B)
    g1[1] = (K & 0xFFFF) << 16;                   // tensor_dim0[15:0] @bits 63:48
    g1[2] = ((K >> 16) & 0xFFFF)                  // tensor_dim0[31:16]
          | ((M & 0xFFFF) << 16);                 // tensor_dim1[15:0]
    g1[3] = (BK << 16);                           // tile_dim0 = 32 @bits 127:112
    g1[4] = BM;                                   // tile_dim1 = 128
    g1[5] = K;                                    // tensor_dim0_stride[31:0]
    g1[6] = 0;                                    // stride hi / dim1_stride lo
    g1[7] = 0;

    const i32x4 gz = (i32x4)0;
#if defined(__clang_major__) && (__clang_major__ >= 23)
    __builtin_amdgcn_tensor_load_to_lds(g0, g1, gz, gz, (i32x8)0, 0);
#else
    __builtin_amdgcn_tensor_load_to_lds(g0, g1, gz, gz, 0);
#endif
}

// ---------------------------------------------------------------------------
// GEMM: D[M,N] = A[M,K](f16) * B[K,N](f32 -> f16) + bias[N]
// Block tile 128x128, K-step 32, ping-pong LDS; 8 waves, each 32x64 strip
// (8 WMMA per K-step). A tiles: TDM async DMA. B tiles: lane-coalesced
// strided f32 loads -> cvt f16 -> 2x ds_store_b128 (n-major => frag loads
// are 2x ds_load_b128). GUARD=false instantiations (all n-blocks fully in
// range) carry zero bounds logic in the hot loop and epilogue.
// ---------------------------------------------------------------------------
template <int N, int K, bool GUARD>
__global__ __launch_bounds__(256)
void snn_gemm_f16(const _Float16* __restrict__ A, const float* __restrict__ B,
                  const float* __restrict__ bias, float* __restrict__ D,
                  int M, int bnOff) {
    __shared__ __align__(16) _Float16 sA [2][BM][SAK];  // 2 x 10 KB (TDM dest)
    __shared__ __align__(16) _Float16 sBt[2][BN][SBK];  // 2 x 10 KB (transposed)

    const int tid  = threadIdx.x;
    const int lane = tid & 31;
    const int wid  = tid >> 5;
    const int wtm  = wid & 3;          // 32-row strip
    const int wtn  = wid >> 2;         // 64-col strip
    const int bm   = blockIdx.x * BM;
    const int bn   = blockIdx.y * BN + bnOff;

    // ---- B staging mapping: lane-consecutive n => coalesced 128 B loads ----
    const int bnl = tid & 127;         // local n
    const int bkh = (tid >> 7) << 4;   // 0 or 16 (K half)
    const int gn  = bn + bnl;
    const bool bOk = GUARD ? (gn < N) : true;

    F16x8 bS0, bS1;
    auto loadB = [&](int k0) {
        if (bOk) {
            const float* gb = B + (size_t)(k0 + bkh) * N + gn;
#pragma unroll
            for (int i = 0; i < 8; ++i) {
                bS0.h[i] = (_Float16)gb[(size_t)i * N];
                bS1.h[i] = (_Float16)gb[(size_t)(i + 8) * N];
            }
        } else {
#pragma unroll
            for (int i = 0; i < 8; ++i) { bS0.h[i] = (_Float16)0.f; bS1.h[i] = (_Float16)0.f; }
        }
    };
    auto storeB = [&](int buf) {
        *(float4*)(&sBt[buf][bnl][bkh])     = bS0.f4;   // 16 contiguous f16
        *(float4*)(&sBt[buf][bnl][bkh + 8]) = bS1.f4;
    };

    v8f acc[2][4];
#pragma unroll
    for (int i = 0; i < 2; ++i)
#pragma unroll
        for (int j = 0; j < 4; ++j)
            acc[i][j] = (v8f){0.f, 0.f, 0.f, 0.f, 0.f, 0.f, 0.f, 0.f};

    // ---- prologue: stage tile 0 ----
    if (wid == 0)
        tdm_load_a_tile<K>(A + (size_t)bm * K, &sA[0][0][0], M);
    loadB(0);
    storeB(0);
    if (wid == 0)
        __builtin_amdgcn_s_wait_tensorcnt(0);
    __syncthreads();

    int cur = 0;
    for (int k0 = 0; k0 < K; k0 += BK) {
        const bool hasNext = (k0 + BK) < K;
        if (hasNext) {
            if (wid == 0)   // one DMA per workgroup; TDM ignores EXEC
                tdm_load_a_tile<K>(A + (size_t)bm * K + (k0 + BK), &sA[cur ^ 1][0][0], M);
            loadB(k0 + BK);                // overlap global latency w/ WMMA
        }

        // ---- A fragments: lanes 0-15 kb=0 (K 0..7,16..23), lanes 16-31
        //      kb=8 (K 8..15,24..31); 2x ds_load_b128 each ----
        F16x16 af[2];
#pragma unroll
        for (int i = 0; i < 2; ++i) {
            const int r  = wtm * 32 + i * 16 + (lane & 15);
            const int kb = (lane >> 4) << 3;
            const _Float16* p = &sA[cur][r][kb];
            af[i].f4[0] = *(const float4*)p;          // K = kb + 0..7
            af[i].f4[1] = *(const float4*)(p + 16);   // K = kb + 16..23
        }
        // ---- B fragments: lane n = n0+lane%16, K = kb2..kb2+7, kb2+16..kb2+23 ----
#pragma unroll
        for (int j = 0; j < 4; ++j) {
            F16x16 bf;
            const int n   = wtn * 64 + j * 16 + (lane & 15);
            const int kb2 = (lane >> 4) << 3;
            const _Float16* p = &sBt[cur][n][kb2];
            bf.f4[0] = *(const float4*)p;             // K = kb2 + 0..7
            bf.f4[1] = *(const float4*)(p + 16);      // K = kb2 + 16..23
#pragma unroll
            for (int i = 0; i < 2; ++i)
                acc[i][j] = __builtin_amdgcn_wmma_f32_16x16x32_f16(
                    false, af[i].v, false, bf.v, (short)0, acc[i][j], false, false);
        }

        if (hasNext) storeB(cur ^ 1);
        if (wid == 0 && hasNext)
            __builtin_amdgcn_s_wait_tensorcnt(0);     // next A tile landed
        __syncthreads();                              // publish next buffer
        cur ^= 1;
    }

    // ---- store D: vgpr r -> row (+8 for hi lanes), col = lane%16 ----
#pragma unroll
    for (int i = 0; i < 2; ++i) {
        const int mbase = bm + wtm * 32 + i * 16 + ((lane >> 4) << 3);
#pragma unroll
        for (int j = 0; j < 4; ++j) {
            const int n = bn + wtn * 64 + j * 16 + (lane & 15);
            if (!GUARD || n < N) {
                const float bv = bias[n];
#pragma unroll
                for (int r = 0; r < 8; ++r)
                    D[(size_t)(mbase + r) * N + n] = acc[i][j][r] + bv;
            }
        }
    }
}

// ---------------------------------------------------------------------------
// Embedding gather + f32->f16 convert: xh[row, d] = emb[ids[row], d]
// ---------------------------------------------------------------------------
__global__ __launch_bounds__(256)
void snn_gather_embed(const int* __restrict__ ids, const float* __restrict__ emb,
                      _Float16* __restrict__ xh, int NT, int Dm) {
    const int q = Dm >> 2;
    const int idx = blockIdx.x * blockDim.x + threadIdx.x;
    if (idx >= NT * q) return;
    const int row = idx / q;
    const int c4  = (idx - row * q) << 2;
    const int tok = ids[row];
    const float4 v = *(const float4*)(emb + (size_t)tok * Dm + c4);
    _Float16* o = xh + (size_t)row * Dm + c4;
    o[0] = (_Float16)v.x; o[1] = (_Float16)v.y;
    o[2] = (_Float16)v.z; o[3] = (_Float16)v.w;
}

// ---------------------------------------------------------------------------
// Adaptive-LIF scan over T: one thread per (b, h) chain. Spikes are exactly
// 0.0/1.0 -> f16 output is lossless and feeds the fc2 WMMA A operand.
// ---------------------------------------------------------------------------
__global__ __launch_bounds__(256)
void snn_lif_scan(const float* __restrict__ pre, _Float16* __restrict__ spk,
                  int Bn, int T, int H) {
    const int idx = blockIdx.x * blockDim.x + threadIdx.x;
    if (idx >= Bn * H) return;
    const int b = idx / H;
    const int h = idx - b * H;
    const float* p = pre + (size_t)b * T * H + h;
    _Float16*   o = spk + (size_t)b * T * H + h;
    float mem = 0.f, adapt = 0.f;
    for (int t = 0; t < T; ++t) {
        mem = mem * 0.9f + p[(size_t)t * H];
        const float thr = 1.0f + adapt;
        const float s   = (mem > thr) ? 1.0f : 0.0f;
        mem   = mem * (1.0f - s);
        adapt = adapt * 0.9f + 0.1f * s;
        o[(size_t)t * H] = (_Float16)s;
    }
}

// ---------------------------------------------------------------------------
extern "C" void kernel_launch(void* const* d_in, const int* in_sizes, int n_in,
                              void* d_out, int out_size, void* d_ws, size_t ws_size,
                              hipStream_t stream) {
    (void)in_sizes; (void)n_in; (void)out_size; (void)ws_size;

    const int*   ids = (const int*)  d_in[0];
    const float* emb = (const float*)d_in[1];
    const float* w1  = (const float*)d_in[2];
    const float* b1  = (const float*)d_in[3];
    const float* w2  = (const float*)d_in[4];
    const float* b2  = (const float*)d_in[5];
    float* out = (float*)d_out;

    const int Bsz = 8, T = 256, Dm = 512, H = 1024, V = 50257;
    const int MT = Bsz * T;                        // 2048
    const int nFull = V / BN;                      // 392 guard-free n-blocks

    char* ws = (char*)d_ws;
    _Float16* xh  = (_Float16*)(ws);               //  2 MB  [MT, Dm] f16
    float*    pre = (float*)   (ws + (4u << 20));  //  8.4MB [MT, H]  f32
    _Float16* spk = (_Float16*)(ws + (16u << 20)); //  4.2MB [MT, H]  f16

    // 1) embedding gather + f16 convert
    {
        const int n = MT * (Dm / 4);
        snn_gather_embed<<<(n + 255) / 256, 256, 0, stream>>>(ids, emb, xh, MT, Dm);
    }
    // 2) fc1: pre = xh @ w1 + b1   (M=2048, N=1024, K=512) -- guard-free
    {
        dim3 grid(MT / BM, H / BN);
        snn_gemm_f16<1024, 512, false><<<grid, 256, 0, stream>>>(xh, w1, b1, pre, MT, 0);
    }
    // 3) LIF scan -> spikes (f16 0/1)
    {
        const int n = Bsz * H;
        snn_lif_scan<<<(n + 255) / 256, 256, 0, stream>>>(pre, spk, Bsz, T, H);
    }
    // 4) fc2: logits = spk @ w2 + b2  (M=2048, N=50257, K=1024)
    //    grid.x = m-blocks (fastest): all 16 m-blocks of one n-column reuse the
    //    512 KB w2 slice out of the 192 MB L2 -> w2 streams from HBM ~once.
    //    Interior 392 n-blocks are guard-free; the last 81 columns take the
    //    guarded edge instantiation.
    {
        dim3 gridI(MT / BM, nFull);
        snn_gemm_f16<50257, 1024, false><<<gridI, 256, 0, stream>>>(spk, w2, b2, out, MT, 0);
        dim3 gridE(MT / BM, 1);
        snn_gemm_f16<50257, 1024, true><<<gridE, 256, 0, stream>>>(spk, w2, b2, out, MT, nFull * BN);
    }
}
